// ConditionalRandomFieldBiRNN_30777735643368
// MI455X (gfx1250) — matrix operand order repaired and checked by reading, not compiled, not run on hardware
//
#include <hip/hip_runtime.h>

// ---------------------------------------------------------------------------
// CRF/HMM forward-algorithm log-likelihood on MI455X (gfx1250, wave32, WMMA)
//
// Kernel 0: one-time RNE pack of ThetaB -> bf16 in exact WMMA-B lane layout
// Kernel 1: gathered GEMM  emit[bt][k] = exp( E[words[bt]] . ThetaB[k] )
//           via v_wmma_f32_16x16x32_bf16; A packed with v_perm_b32,
//           B operands streamed from the 64KB pre-packed table (L2-hot)
// Kernel 2: batched scaled forward recursion, 16 chains per wave,
//           per step 64x v_wmma_f32_16x16x4_f32 ([16x64] @ [64x64]);
//           per-step emissions fetched by the Tensor Data Mover
//           (tensor_load_to_lds, double-buffered, s_wait_tensorcnt)
// ---------------------------------------------------------------------------

typedef __attribute__((ext_vector_type(16))) __bf16  v16bf;
typedef __attribute__((ext_vector_type(8)))  float   v8f;
typedef __attribute__((ext_vector_type(2)))  float   v2f;
typedef __attribute__((ext_vector_type(4)))  unsigned v4u;
typedef __attribute__((ext_vector_type(8)))  int      v8i;
typedef __attribute__((ext_vector_type(4)))  int      v4i;

union BF16x16 { v16bf v; unsigned d[8]; uint4 q[2]; };

#define KTAGS 64
#define DDIM  512
#define TLEN  128
#define BOS_T 63
#define EOS_T 62

__device__ __forceinline__ unsigned fbits(float f) {
  union { float f; unsigned u; } c; c.f = f;
  return c.u;
}
__device__ __forceinline__ unsigned short f2bf_rne(float f) {
  unsigned u = fbits(f);
  unsigned r = u + 0x7FFFu + ((u >> 16) & 1u);
  return (unsigned short)(r >> 16);
}
// pack two f32 -> (bf16(f0) | bf16(f1)<<16) via one v_perm_b32 (truncate)
__device__ __forceinline__ unsigned pk_bf16(float f0, float f1) {
  return __builtin_amdgcn_perm(fbits(f1), fbits(f0), 0x07060302u);
}
// generic pointer (to a __shared__ object) -> raw 32-bit LDS byte offset
#define LDS_OFF(p) ((unsigned)(unsigned long long)(__attribute__((address_space(3))) const void*)(p))

// ---------------------------------------------------------------------------
// Kernel 0: pack ThetaB into WMMA-B register layout, bf16 RNE.
// PB[((kk/32)*4 + nt)*32 + lane] = 16 bf16:
//   col = nt*16 + (lane&15), K = kk + (lane>=16 ? 16 : 0) + 0..15
// ---------------------------------------------------------------------------
__global__ __launch_bounds__(256) void pack_thetaB_kernel(
    const float* __restrict__ ThetaB, unsigned* __restrict__ PB)
{
  const int tid  = blockIdx.x * 256 + threadIdx.x;  // 0..2047
  const int lane = tid & 31;
  const int nt   = (tid >> 5) & 3;
  const int kk16 = tid >> 7;                        // 0..15
  const int lidx = lane & 15;
  const int half = lane >> 4;

  const float* __restrict__ tb =
      ThetaB + (size_t)(nt * 16 + lidx) * DDIM + kk16 * 32 + half * 16;

  unsigned d[8];
#pragma unroll
  for (int i = 0; i < 8; ++i) {
    unsigned lo = f2bf_rne(tb[2 * i]);
    unsigned hi = f2bf_rne(tb[2 * i + 1]);
    d[i] = lo | (hi << 16);
  }
  uint4* dst = (uint4*)(PB + (size_t)tid * 8);
  dst[0] = make_uint4(d[0], d[1], d[2], d[3]);
  dst[1] = make_uint4(d[4], d[5], d[6], d[7]);
}

// ---------------------------------------------------------------------------
// Kernel 1: emission GEMM. 512 blocks x 256 threads (8 waves);
// each wave: 16 word-positions x 64 tags, K-loop 16 x (K=32 bf16 WMMA).
// ---------------------------------------------------------------------------
__global__ __launch_bounds__(256) void emit_gemm_kernel(
    const unsigned* __restrict__ PB,    // pre-packed ThetaB (64KB, L2-hot)
    const float*    __restrict__ E,     // [100000][512]
    const int*      __restrict__ words, // [65536]
    float*          __restrict__ emit)  // [65536][64]
{
  const int lane = threadIdx.x & 31;
  const int wave = threadIdx.x >> 5;
  const int tile = blockIdx.x * 8 + wave;     // 4096 tiles of 16 word positions
  const int m0   = tile * 16;
  const int half = lane >> 4;
  const int lidx = lane & 15;

  const int row = words[m0 + lidx];
  const float* __restrict__ Erow = E + (size_t)row * DDIM;
  const uint4* __restrict__ PBq  = (const uint4*)PB;

  const int koffA = half ? 16 : 0;

  v8f acc[4];
#pragma unroll
  for (int nt = 0; nt < 4; ++nt) acc[nt] = (v8f)0.0f;

  for (int kk = 0; kk < DDIM; kk += 32) {
    if (kk + 32 < DDIM) __builtin_prefetch(Erow + kk + 32 + koffA, 0, 1);

    // ---- A operand: gathered E row chunk, v_perm_b32 packing ----
    BF16x16 A;
    {
      const float4* p0 = (const float4*)(Erow + kk + koffA);
      const float4* p1 = (const float4*)(Erow + kk + koffA + 8);
      float4 a0 = p0[0], a1 = p0[1];
      float4 b0 = p1[0], b1 = p1[1];
      A.d[0] = pk_bf16(a0.x, a0.y);  A.d[1] = pk_bf16(a0.z, a0.w);
      A.d[2] = pk_bf16(a1.x, a1.y);  A.d[3] = pk_bf16(a1.z, a1.w);
      A.d[4] = pk_bf16(b0.x, b0.y);  A.d[5] = pk_bf16(b0.z, b0.w);
      A.d[6] = pk_bf16(b1.x, b1.y);  A.d[7] = pk_bf16(b1.z, b1.w);
    }

#pragma unroll
    for (int nt = 0; nt < 4; ++nt) {
      const int idx = (((kk >> 5) * 4) + nt) * 32 + lane;
      BF16x16 Bv;
      Bv.q[0] = PBq[2 * idx];
      Bv.q[1] = PBq[2 * idx + 1];
      acc[nt] = __builtin_amdgcn_wmma_f32_16x16x32_bf16(
          false, A.v, false, Bv.v, (short)0, acc[nt], false, false);
    }
  }

  // D layout: (lane,v) -> word m0 + v + 8*half, tag nt*16 + lidx
#pragma unroll
  for (int nt = 0; nt < 4; ++nt) {
    const int jj = nt * 16 + lidx;
#pragma unroll
    for (int v = 0; v < 8; ++v) {
      const int m = v + half * 8;
      float val = (jj >= EOS_T) ? 1e-45f : __expf(acc[nt][v]);
      emit[(size_t)(m0 + m) * KTAGS + jj] = val;
    }
  }
}

// ---------------------------------------------------------------------------
// TDM: load an 8-row x 64-f32 tile (row stride 8192 f32) into LDS.
// Descriptor per CDNA5 ISA; 6-arg builtin form
// (this toolchain: (u32x4 g0, i32x8 g1, i32x4 g2, i32x4 g3, i32x8 g4, cpol)).
// ---------------------------------------------------------------------------
__device__ __forceinline__ void tdm_load_emit_rows(unsigned long long gaddr,
                                                   unsigned lds_addr) {
  v4u g0;
  g0.x = 1u;                                       // count=1, user descriptor
  g0.y = lds_addr;                                 // LDS byte address
  g0.z = (unsigned)gaddr;                          // global_addr[31:0]
  g0.w = (unsigned)(gaddr >> 32) | (2u << 30);     // global_addr[56:32] | type=2
  v8i g1;
  g1[0] = 0x00020000;                              // data_size = 4B
  g1[1] = (64 << 16);                              // tensor_dim0 = 64
  g1[2] = (8 << 16);                               // tensor_dim1 = 8
  g1[3] = (64 << 16);                              // tile_dim0 = 64
  g1[4] = 8;                                       // tile_dim1 = 8, tile_dim2 = 0
  g1[5] = TLEN * KTAGS;                            // tensor_dim0_stride = 8192
  g1[6] = 0;
  g1[7] = 0;
  v4i z4 = {0, 0, 0, 0};                           // groups 2/3 unused (2D tile)
  v8i z8 = {0, 0, 0, 0, 0, 0, 0, 0};
  __builtin_amdgcn_tensor_load_to_lds(g0, g1, z4, z4, z8, 0);
}

// ---------------------------------------------------------------------------
// Kernel 2: forward recursion. 64 blocks x 32 threads (1 wave = 16 chains:
// 8 sequences x {tagged, unsupervised}).
// ---------------------------------------------------------------------------
__global__ __launch_bounds__(32) void forward_kernel(
    const float* __restrict__ WA,     // [64][64]
    const float* __restrict__ emit,   // [65536][64] (L2-resident, 16.8MB)
    const int*   __restrict__ tags,   // [512][128]
    float*       __restrict__ out)    // [512]
{
  __shared__ __align__(16) float2 Atp[4][16][32];      // WMMA-A tiles of A^T, 16KB
  __shared__ __align__(16) float  alphaT[2][16][68];   // padded, double-buffered
  __shared__ __align__(16) float  eosA[64];            // A[:,EOS]
  __shared__ __align__(16) float  stage[2][8][64];     // TDM emission tiles, 4KB
  __shared__ __align__(16) int    tagS[8 * TLEN];      // staged tags, 4KB

  const int lane    = threadIdx.x;
  const int half    = lane >> 4;
  const int c       = lane & 15;         // chain id
  const int s       = c & 7;             // sequence within the 8-seq group
  const bool tagged = (c < 8);
  const int b0      = blockIdx.x * 8;
  const int b       = b0 + s;
  const int koff    = half ? 2 : 0;

  // ---- pack A^T = exp(WA)^T (BOS column zeroed) into WMMA-A layout ----
#pragma unroll
  for (int jt = 0; jt < 4; ++jt) {
    const int j = jt * 16 + c;
#pragma unroll
    for (int kt = 0; kt < 16; ++kt) {
      const int i0 = kt * 4 + koff;
      float x0 = (j == BOS_T) ? 0.0f : __expf(WA[i0 * KTAGS + j]);
      float x1 = (j == BOS_T) ? 0.0f : __expf(WA[(i0 + 1) * KTAGS + j]);
      Atp[jt][kt][lane] = make_float2(x0, x1);
    }
  }
  eosA[lane]      = __expf(WA[lane * KTAGS + EOS_T]);
  eosA[lane + 32] = __expf(WA[(lane + 32) * KTAGS + EOS_T]);

  // stage tags for the 8 sequences
  for (int idx = lane; idx < 8 * TLEN; idx += 32)
    tagS[idx] = tags[(b0 + (idx >> 7)) * TLEN + (idx & (TLEN - 1))];

  // ---- alpha0: 1 at BOS. D layout: regD[jt][v] <-> i = jt*16 + v + 8*half ----
  v8f regD[4];
#pragma unroll
  for (int jt = 0; jt < 4; ++jt) regD[jt] = (v8f)0.0f;
  regD[3][7] = half ? 1.0f : 0.0f;       // i = 48+7+8 = 63 = BOS_T
  for (int r = 0; r < 32; ++r) {
    const int i = half * 32 + r;
    alphaT[0][c][i] = (i == BOS_T) ? 1.0f : 0.0f;
  }

  // prime the TDM pipeline: emissions for t=1 into stage[1]
  const unsigned long long emitBase = (unsigned long long)(size_t)emit;
  tdm_load_emit_rows(emitBase + ((unsigned long long)b0 * TLEN + 1) * KTAGS * 4,
                     LDS_OFF(&stage[1][0][0]));

  float logsum = 0.0f;
  int cur = 0;

  for (int t = 1; t <= TLEN - 2; ++t) {
    // prefetch next step's emissions via TDM while we compute this step
    if (t < TLEN - 2) {
      tdm_load_emit_rows(
          emitBase + ((unsigned long long)b0 * TLEN + (t + 1)) * KTAGS * 4,
          LDS_OFF(&stage[(t + 1) & 1][0][0]));
      __builtin_amdgcn_s_wait_tensorcnt(1);   // current step's tile is ready
    } else {
      __builtin_amdgcn_s_wait_tensorcnt(0);
    }

    // scaling factor sc = max_i alpha[i][c]
    float mx = 0.0f;
#pragma unroll
    for (int jt = 0; jt < 4; ++jt)
#pragma unroll
      for (int v = 0; v < 8; ++v) mx = fmaxf(mx, regD[jt][v]);
    mx = fmaxf(mx, __shfl_xor(mx, 16));
    logsum += __logf(mx);
    const float inv = 1.0f / mx;

    const int tagv = tagS[s * TLEN + t];
    const float* __restrict__ ep = &stage[t & 1][s][0];

    const int nxt = cur ^ 1;
#pragma unroll
    for (int jt = 0; jt < 4; ++jt) {
      v8f acc = (v8f)0.0f;
#pragma unroll
      for (int kt = 0; kt < 16; ++kt) {
        const float2 a2 = Atp[jt][kt][lane];                   // ds_load_b64
        const int i0 = kt * 4 + koff;
        const float2 b2 = *(const float2*)&alphaT[cur][c][i0]; // ds_load_b64
        v2f av; av.x = a2.x;       av.y = a2.y;
        v2f bv; bv.x = b2.x * inv; bv.y = b2.y * inv;
        acc = __builtin_amdgcn_wmma_f32_16x16x4_f32(
            false, av, false, bv, (short)0, acc, false, false);
      }
      // elementwise: x emit x (one-hot tag for supervised chains)
      const int j0 = jt * 16 + half * 8;
#pragma unroll
      for (int v = 0; v < 8; ++v) {
        const int j = j0 + v;
        const float e = ep[j];                                 // LDS (TDM tile)
        const float f = tagged ? ((j == tagv) ? 1.0f : 0.0f) : 1.0f;
        const float nv = acc[v] * e * f;
        regD[jt][v] = nv;
        alphaT[nxt][c][j] = nv;
      }
    }
    cur = nxt;
  }

  // ---- termination: val = (alpha/sc) . A[:,EOS] ----
  float mx = 0.0f;
#pragma unroll
  for (int jt = 0; jt < 4; ++jt)
#pragma unroll
    for (int v = 0; v < 8; ++v) mx = fmaxf(mx, regD[jt][v]);
  mx = fmaxf(mx, __shfl_xor(mx, 16));
  const float inv = 1.0f / mx;

  float part = 0.0f;
#pragma unroll
  for (int jt = 0; jt < 4; ++jt)
#pragma unroll
    for (int v = 0; v < 8; ++v) {
      const int i = jt * 16 + half * 8 + v;
      part += regD[jt][v] * inv * eosA[i];
    }
  const float val  = part + __shfl_xor(part, 16);
  const float logZ = __logf(val) + logsum + __logf(mx);

  // out[b] = logZ_tagged - logZ_unsup  (chain c vs chain c+8)
  const float other = __shfl_xor(logZ, 8);
  if (lane < 8) out[b] = logZ - other;
}

// ---------------------------------------------------------------------------
extern "C" void kernel_launch(void* const* d_in, const int* in_sizes, int n_in,
                              void* d_out, int out_size, void* d_ws, size_t ws_size,
                              hipStream_t stream) {
  const float* WA     = (const float*)d_in[0];   // [64,64]
  const float* ThetaB = (const float*)d_in[1];   // [64,512]
  const float* E      = (const float*)d_in[2];   // [100000,512]
  const int*   words  = (const int*)  d_in[3];   // [512,128]
  const int*   tags   = (const int*)  d_in[4];   // [512,128]
  float*       out    = (float*)d_out;           // [512]

  float*    emit = (float*)d_ws;                          // 16.8 MB scratch
  unsigned* PB   = (unsigned*)((char*)d_ws + (size_t)65536 * KTAGS * 4); // 64 KB

  hipLaunchKernelGGL(pack_thetaB_kernel, dim3(8), dim3(256), 0, stream,
                     ThetaB, PB);
  hipLaunchKernelGGL(emit_gemm_kernel, dim3(512), dim3(256), 0, stream,
                     PB, E, words, emit);
  hipLaunchKernelGGL(forward_kernel, dim3(64), dim3(32), 0, stream,
                     WA, emit, tags, out);
}
